// VanillaLSTM_71708773974722
// MI455X (gfx1250) — compile-verified, matrix operand order checked by reading, not run on hardware
//
#include <hip/hip_runtime.h>
#include <hip/hip_bf16.h>

#define BATCH 4096
#define EMBD  64
#define HID   1024
#define KTOT  1088   // EMBD + HID
#define NGATE 4

#define KC    32     // K chunk per stage
#define MB    128    // batch rows per block
#define NB    32     // cols (per gate) per block
#define LDSW  40     // padded LDS row stride in bf16 elements (80 B) -> conflict-free

typedef __attribute__((ext_vector_type(16))) __bf16 v16bf;
typedef __attribute__((ext_vector_type(8)))  float  v8f;
typedef __attribute__((ext_vector_type(4)))  float  v4f;

__device__ __forceinline__ float sigm(float x) { return 1.0f / (1.0f + __expf(-x)); }
// branch-free tanh via exp: tanh(x) = 2*sigmoid(2x) - 1
__device__ __forceinline__ float fast_tanh(float x) { return __builtin_fmaf(2.0f, sigm(2.0f * x), -1.0f); }

// One 16x32 bf16 WMMA fragment: lane l<16 -> row l, K={0..7,16..23};
// lane>=16 -> row l-16, K={8..15,24..31}. p points at (row, khalf).
__device__ __forceinline__ v16bf load_frag(const __bf16* p) {
    v16bf v;
    *reinterpret_cast<v4f*>(&v)       = *reinterpret_cast<const v4f*>(p);
    *(reinterpret_cast<v4f*>(&v) + 1) = *reinterpret_cast<const v4f*>(p + 16);
    return v;
}

// CDNA5 async copy: 16 bytes per lane, global -> LDS, tracked by ASYNCcnt.
__device__ __forceinline__ void async_cp16(uint32_t lds_byte, const __bf16* g) {
    asm volatile("global_load_async_to_lds_b128 %0, %1, off"
                 :: "v"(lds_byte), "v"(g) : "memory");
}

// ---------------------------------------------------------------------------
// Fused gates GEMM + LSTM cell, LDS double-buffered via async-to-LDS.
// Block: 256 thr = 8 waves = 128 batch rows x 32 cols x 4 gates.
// Wave (wr 0..3, wc 0..1): 32 rows x 16 cols x 4 gates -> 8 accumulators.
// Grid: (HID/NB, BATCH/MB) = (32, 32).
// ---------------------------------------------------------------------------
__global__ __launch_bounds__(256)
void lstm_gates_kernel(const __bf16* __restrict__ A,     // [BATCH, KTOT] = [emb | h]
                       __bf16*       __restrict__ Anext, // h region (cols 64..) written
                       const __bf16* __restrict__ W,     // [4*HID, KTOT] row-major
                       const float*  __restrict__ bias,  // [4*HID]
                       float*        __restrict__ c,     // [BATCH, HID]
                       float*        __restrict__ h)     // [BATCH, HID]
{
    __shared__ __bf16 As[2][MB * LDSW];   // 128 x 32 (+pad) per buffer
    __shared__ __bf16 Bs[2][MB * LDSW];   // (4 gates x 32 cols) x 32 (+pad)

    const int tid   = threadIdx.x;
    const int wave  = tid >> 5;
    const int lane  = tid & 31;
    const int wr    = wave & 3;            // 32-row group
    const int wc    = wave >> 2;           // 16-col group
    const int m0b   = blockIdx.y * MB;
    const int n0b   = blockIdx.x * NB;
    const int lrow  = lane & 15;
    const int khalf = (lane >> 4) << 3;    // 0 or 8

    const __bf16* gA = A + (size_t)m0b * KTOT;

    uint32_t asb[2], bsb[2];
    asb[0] = (uint32_t)(size_t)&As[0][0];
    asb[1] = (uint32_t)(size_t)&As[1][0];
    bsb[0] = (uint32_t)(size_t)&Bs[0][0];
    bsb[1] = (uint32_t)(size_t)&Bs[1][0];

    // Each thread stages 2x16B of A and 2x16B of B per chunk (512 16B pieces each).
    const int q0 = tid, q1 = tid + 256;

    auto stage = [&](int k0, int buf) {
        int row = q0 >> 2, kq = (q0 & 3) << 3;
        async_cp16(asb[buf] + row * (LDSW * 2) + (kq << 1), gA + (size_t)row * KTOT + k0 + kq);
        row = q1 >> 2; kq = (q1 & 3) << 3;
        async_cp16(asb[buf] + row * (LDSW * 2) + (kq << 1), gA + (size_t)row * KTOT + k0 + kq);

        int r2 = q0 >> 2; kq = (q0 & 3) << 3;
        int g = r2 >> 5, cc = r2 & 31;
        async_cp16(bsb[buf] + r2 * (LDSW * 2) + (kq << 1),
                   W + ((size_t)g * HID + n0b + cc) * KTOT + k0 + kq);
        r2 = q1 >> 2; kq = (q1 & 3) << 3;
        g = r2 >> 5; cc = r2 & 31;
        async_cp16(bsb[buf] + r2 * (LDSW * 2) + (kq << 1),
                   W + ((size_t)g * HID + n0b + cc) * KTOT + k0 + kq);
    };

    v8f acc[8] = {};  // acc[g*2 + rowfrag]

    const int a0off = (wr * 32 + lrow) * LDSW + khalf;
    const int a1off = a0off + 16 * LDSW;
    const int boff  = (wc * 16 + lrow) * LDSW + khalf;

    stage(0, 0);

    constexpr int NCHUNK = KTOT / KC;  // 34
#pragma unroll 2
    for (int i = 0; i < NCHUNK; ++i) {
        asm volatile("s_wait_asynccnt 0x0" ::: "memory");  // my stage(i) done
        __syncthreads();                                   // everyone's stage(i) done,
                                                           // everyone finished reading buf i-1
        if (i + 1 < NCHUNK) stage((i + 1) * KC, (i + 1) & 1);

        const __bf16* as = &As[i & 1][0];
        const __bf16* bs = &Bs[i & 1][0];
        v16bf a0 = load_frag(as + a0off);
        v16bf a1 = load_frag(as + a1off);
#pragma unroll
        for (int g = 0; g < 4; ++g) {
            v16bf wf = load_frag(bs + boff + g * 32 * LDSW);
            acc[g * 2 + 0] = __builtin_amdgcn_wmma_f32_16x16x32_bf16(
                false, a0, false, wf, (short)0, acc[g * 2 + 0], false, false);
            acc[g * 2 + 1] = __builtin_amdgcn_wmma_f32_16x16x32_bf16(
                false, a1, false, wf, (short)0, acc[g * 2 + 1], false, false);
        }
    }

    // Epilogue: D layout -> lane l<16: N=l, M=r; lane>=16: N=l-16, M=r+8.
    const int   n   = n0b + wc * 16 + lrow;
    const float bbi = bias[0 * HID + n];
    const float bbf = bias[1 * HID + n];
    const float bbg = bias[2 * HID + n];
    const float bbo = bias[3 * HID + n];
#pragma unroll
    for (int rf = 0; rf < 2; ++rf) {
#pragma unroll
        for (int r = 0; r < 8; ++r) {
            const int    m   = m0b + wr * 32 + rf * 16 + r + ((lane >> 4) << 3);
            const size_t idx = (size_t)m * HID + n;
            const float iv = sigm(acc[0 + rf][r] + bbi);
            const float fv = sigm(acc[2 + rf][r] + bbf);
            const float gv = fast_tanh(acc[4 + rf][r] + bbg);
            const float ov = sigm(acc[6 + rf][r] + bbo);
            const float cn = fv * c[idx] + iv * gv;
            const float hn = ov * fast_tanh(cn);
            c[idx] = cn;
            h[idx] = hn;
            Anext[(size_t)m * KTOT + EMBD + n] = (__bf16)hn;
        }
    }
}

// ---------------------------------------------------------------------------
// Embedding: emb = relu(d @ W_emb^T + b_emb) -> bf16 into A[:, 0:64].
// mode 0: d = s1 - s0 ([B,2] diffs); mode 1: d = s0[:, :2] of [B,5].
// ---------------------------------------------------------------------------
__global__ void emb_kernel(const float* __restrict__ s0,
                           const float* __restrict__ s1,
                           const float* __restrict__ Wemb,
                           const float* __restrict__ bemb,
                           __bf16*      __restrict__ A,
                           int mode)
{
    const int idx = blockIdx.x * blockDim.x + threadIdx.x;
    if (idx >= BATCH * EMBD) return;
    const int b = idx >> 6;
    const int e = idx & 63;
    float d0, d1;
    if (mode == 0) {
        d0 = s1[b * 2 + 0] - s0[b * 2 + 0];
        d1 = s1[b * 2 + 1] - s0[b * 2 + 1];
    } else {
        d0 = s0[b * 5 + 0];
        d1 = s0[b * 5 + 1];
    }
    float v = d0 * Wemb[e * 2 + 0] + d1 * Wemb[e * 2 + 1] + bemb[e];
    v = v > 0.0f ? v : 0.0f;
    A[(size_t)b * KTOT + e] = (__bf16)v;
}

// ---------------------------------------------------------------------------
// Output projection: normal = h @ W_out^T + b_out. One wave per batch row.
// ---------------------------------------------------------------------------
__global__ void outproj_kernel(const float* __restrict__ h,
                               const float* __restrict__ Wout,
                               const float* __restrict__ bout,
                               float*       __restrict__ out)
{
    const int wave = threadIdx.x >> 5;
    const int lane = threadIdx.x & 31;
    const int b    = blockIdx.x * 8 + wave;
    const float* hb = h + (size_t)b * HID;
    float a0 = 0.f, a1 = 0.f, a2 = 0.f, a3 = 0.f, a4 = 0.f;
    for (int k = lane; k < HID; k += 32) {
        const float hv = hb[k];
        a0 += hv * Wout[0 * HID + k];
        a1 += hv * Wout[1 * HID + k];
        a2 += hv * Wout[2 * HID + k];
        a3 += hv * Wout[3 * HID + k];
        a4 += hv * Wout[4 * HID + k];
    }
#pragma unroll
    for (int off = 16; off > 0; off >>= 1) {
        a0 += __shfl_down(a0, off, 32);
        a1 += __shfl_down(a1, off, 32);
        a2 += __shfl_down(a2, off, 32);
        a3 += __shfl_down(a3, off, 32);
        a4 += __shfl_down(a4, off, 32);
    }
    if (lane == 0) {
        out[(size_t)b * 5 + 0] = a0 + bout[0];
        out[(size_t)b * 5 + 1] = a1 + bout[1];
        out[(size_t)b * 5 + 2] = a2 + bout[2];
        out[(size_t)b * 5 + 3] = a3 + bout[3];
        out[(size_t)b * 5 + 4] = a4 + bout[4];
    }
}

// ---------------------------------------------------------------------------
// One-time prep kernels.
// ---------------------------------------------------------------------------
__global__ void prep_weights_kernel(const float* __restrict__ Wih,
                                    const float* __restrict__ Whh,
                                    __bf16*      __restrict__ Wcat)
{
    const size_t total  = (size_t)NGATE * HID * KTOT;
    const size_t stride = (size_t)gridDim.x * blockDim.x;
    for (size_t i = (size_t)blockIdx.x * blockDim.x + threadIdx.x; i < total; i += stride) {
        const int n = (int)(i / KTOT);
        const int k = (int)(i % KTOT);
        const float v = (k < EMBD) ? Wih[(size_t)n * EMBD + k]
                                   : Whh[(size_t)n * HID + (k - EMBD)];
        Wcat[i] = (__bf16)v;
    }
}

__global__ void init_state_kernel(const float* __restrict__ bih,
                                  const float* __restrict__ bhh,
                                  float*       __restrict__ bias,
                                  float*       __restrict__ c,
                                  __bf16*      __restrict__ A0)
{
    const size_t total  = (size_t)BATCH * HID;
    const size_t stride = (size_t)gridDim.x * blockDim.x;
    for (size_t i = (size_t)blockIdx.x * blockDim.x + threadIdx.x; i < total; i += stride) {
        c[i] = 0.0f;
        const int b = (int)(i / HID);
        const int n = (int)(i % HID);
        A0[(size_t)b * KTOT + EMBD + n] = (__bf16)0.0f;
        if (i < (size_t)NGATE * HID) bias[i] = bih[i] + bhh[i];
    }
}

extern "C" void kernel_launch(void* const* d_in, const int* in_sizes, int n_in,
                              void* d_out, int out_size, void* d_ws, size_t ws_size,
                              hipStream_t stream) {
    const float* observed = (const float*)d_in[0];
    const float* Wemb     = (const float*)d_in[1];
    const float* bemb     = (const float*)d_in[2];
    const float* Wih      = (const float*)d_in[3];
    const float* bih      = (const float*)d_in[4];
    const float* Whh      = (const float*)d_in[5];
    const float* bhh      = (const float*)d_in[6];
    const float* Wout     = (const float*)d_in[7];
    const float* bout     = (const float*)d_in[8];
    float*       out      = (float*)d_out;

    char*  ws  = (char*)d_ws;
    size_t off = 0;
    auto carve = [&](size_t bytes) -> void* {
        void* p = ws + off;
        off += (bytes + 255) & ~(size_t)255;
        return p;
    };
    __bf16* Wcat = (__bf16*)carve((size_t)NGATE * HID * KTOT * sizeof(__bf16));
    __bf16* A0   = (__bf16*)carve((size_t)BATCH * KTOT * sizeof(__bf16));
    __bf16* A1   = (__bf16*)carve((size_t)BATCH * KTOT * sizeof(__bf16));
    float*  c    = (float*) carve((size_t)BATCH * HID * sizeof(float));
    float*  h    = (float*) carve((size_t)BATCH * HID * sizeof(float));
    float*  bias = (float*) carve((size_t)NGATE * HID * sizeof(float));
    (void)ws_size; (void)in_sizes; (void)n_in;

    const int steps = out_size / (BATCH * 5);  // (SEQ-1) + (n_predict-1) = 19

    prep_weights_kernel<<<4096, 256, 0, stream>>>(Wih, Whh, Wcat);
    init_state_kernel<<<4096, 256, 0, stream>>>(bih, bhh, bias, c, A0);

    const dim3 ggrid(HID / NB, BATCH / MB);   // (32, 32)
    const int  egrid = (BATCH * EMBD) / 256;

    for (int t = 0; t < steps; ++t) {
        __bf16* Acur = (t & 1) ? A1 : A0;
        __bf16* Anxt = (t & 1) ? A0 : A1;
        if (t < 8) {
            emb_kernel<<<egrid, 256, 0, stream>>>(
                observed + (size_t)t * BATCH * 2,
                observed + (size_t)(t + 1) * BATCH * 2,
                Wemb, bemb, Acur, 0);
        } else {
            emb_kernel<<<egrid, 256, 0, stream>>>(
                out + (size_t)(t - 1) * BATCH * 5, nullptr,
                Wemb, bemb, Acur, 1);
        }
        lstm_gates_kernel<<<ggrid, 256, 0, stream>>>(Acur, Anxt, Wcat, bias, c, h);
        outproj_kernel<<<BATCH / 8, 256, 0, stream>>>(h, Wout, bout,
                                                      out + (size_t)t * BATCH * 5);
    }
}
